// PairwiseMSELoss_45664092291536
// MI455X (gfx1250) — matrix operand order repaired and checked by reading, not compiled, not run on hardware
//
#include <hip/hip_runtime.h>

// ---------------------------------------------------------------------------
// PairwiseMSELoss on MI455X (gfx1250, wave32, WMMA + TDM).
//
// out[i] = - (1/D^2) * sum_n (sx_i + sx_n - 2 Gx[i,n]) * (sy_i + sy_n - 2 Gy[i,n])
// Gx = X X^T, Gy = Y Y^T, fused: no BxB matrix in memory. bf16 WMMA, fp32 acc.
// Upper-triangle 128x128 tiles only (t[i,n] symmetric): 528 blocks.
// Main-loop feeds LDS with TENSOR_LOAD_TO_LDS (TDM), double-buffered, with
// hardware LDS row padding (64B data + 16B pad per row via pad_interval/amount).
// ---------------------------------------------------------------------------

typedef __bf16 bf16;
typedef __attribute__((ext_vector_type(16))) __bf16 v16bf;
typedef __attribute__((ext_vector_type(8)))  float  v8f;
typedef unsigned int u32x4 __attribute__((ext_vector_type(4)));
typedef int          i32x4 __attribute__((ext_vector_type(4)));
typedef int          i32x8 __attribute__((ext_vector_type(8)));

#define B_DIM 4096
#define D_DIM 4096
#define TILE  128
#define TK    32
#define LDSW  40            // 32 bf16 + 8 pad -> 80B row stride (20 banks)
#define NTILES (B_DIM / TILE)               // 32
#define NTRI   (NTILES * (NTILES + 1) / 2)  // 528

union FragU { v16bf v; uint4 q[2]; };

// A-matrix 16x32 bf16 fragment (ISA 7.12.2): lane l holds row (l&15);
// lanes 0-15: K = {0..7, 16..23}; lanes 16-31: K = {8..15, 24..31}.
__device__ __forceinline__ v16bf lds_a_frag(const unsigned short t[][LDSW], int r, int h) {
  FragU u;
  u.q[0] = *(const uint4*)&t[r][h * 8];
  u.q[1] = *(const uint4*)&t[r][16 + h * 8];
  return u.v;
}

// B-matrix 32x16 bf16 fragment: lane l holds column (l&15);
// lanes 0-15: K = {0..7, 16..23}; lanes 16-31: K = {8..15, 24..31}.
// For a Gram, B = tile^T, so per-lane K runs are contiguous rows in LDS.
__device__ __forceinline__ v16bf lds_b_frag(const unsigned short t[][LDSW], int c, int h) {
  FragU u;
  u.q[0] = *(const uint4*)&t[c][h * 8];
  u.q[1] = *(const uint4*)&t[c][16 + h * 8];
  return u.v;
}

// ---- TDM: load one 128-row x 32-col bf16 tile (row stride 4096 elems) into
// LDS at lds_off, inserting 16B of padding after every 64B row (-> 80B stride).
__device__ __forceinline__ void tdm_load_tile(const unsigned short* gsrc, unsigned lds_off) {
  const unsigned long long ga = (unsigned long long)(uintptr_t)gsrc;
  u32x4 g0;
  g0[0] = 1u;                                    // count=1, user mode
  g0[1] = lds_off;                               // LDS byte address
  g0[2] = (unsigned)ga;                          // global_addr[31:0]
  g0[3] = (unsigned)((ga >> 32) & 0x01FFFFFFu)   // global_addr[56:32]
          | 0x80000000u;                         // type=2 ("image")
  i32x8 g1;
  g1[0] = (1 << 16)      // data_size = 2 bytes
        | (1 << 20)      // pad_enable
        | (3 << 22)      // pad_interval: every 16 DWORDs (=64B row)
        | (3 << 25);     // pad_amount: 4 DWORDs (=16B)
  g1[1] = (int)(4096u << 16);        // tensor_dim0[15:0]=4096 in [31:16]
  g1[2] = (int)(4096u << 16);        // tensor_dim0 hi=0 | tensor_dim1[15:0]=4096
  g1[3] = (int)(32u << 16);          // tensor_dim1 hi=0 | tile_dim0=32
  g1[4] = 128;                       // tile_dim1=128, tile_dim2=0
  g1[5] = 4096;                      // tensor_dim0_stride[31:0]
  g1[6] = 0;                         // stride hi / dim1_stride lo
  g1[7] = 0;
  const i32x4 z4 = {0, 0, 0, 0};
#if defined(__clang_major__) && (__clang_major__ >= 23)
  const i32x8 z8 = {0, 0, 0, 0, 0, 0, 0, 0};
  __builtin_amdgcn_tensor_load_to_lds(g0, g1, z4, z4, z8, 0);
#else
  __builtin_amdgcn_tensor_load_to_lds(g0, g1, z4, z4, 0);
#endif
}

__device__ __forceinline__ unsigned lds_off_of(const void* p) {
  return (unsigned)(uintptr_t)p;   // low 32 bits of flat LDS address = LDS offset
}

// --------------------------- prep: cvt + row norms --------------------------
__global__ __launch_bounds__(256) void prep_kernel(
    const float* __restrict__ x, const float* __restrict__ y,
    unsigned short* __restrict__ xb, unsigned short* __restrict__ yb,
    float* __restrict__ sqx, float* __restrict__ sqy, float* __restrict__ acc) {
  const int row = blockIdx.x;
  const int t   = threadIdx.x;
  const float* xr = x + (size_t)row * D_DIM;
  const float* yr = y + (size_t)row * D_DIM;
  bf16* xbr = (bf16*)xb + (size_t)row * D_DIM;
  bf16* ybr = (bf16*)yb + (size_t)row * D_DIM;

  float sx = 0.f, sy = 0.f;
  for (int j = t; j < D_DIM; j += 256) {
    float xv = xr[j], yv = yr[j];
    xbr[j] = (bf16)xv;
    ybr[j] = (bf16)yv;
    sx += xv * xv;
    sy += yv * yv;
  }
  __shared__ float rx[256], ry[256];
  rx[t] = sx; ry[t] = sy;
  __syncthreads();
  for (int s = 128; s > 0; s >>= 1) {
    if (t < s) { rx[t] += rx[t + s]; ry[t] += ry[t + s]; }
    __syncthreads();
  }
  if (t == 0) { sqx[row] = rx[0]; sqy[row] = ry[0]; acc[row] = 0.f; }
}

// ------------------- fused dual-Gram WMMA tile + reduce ---------------------
__global__ __launch_bounds__(256) void gram_kernel(
    const unsigned short* __restrict__ xb, const unsigned short* __restrict__ yb,
    const float* __restrict__ sqx, const float* __restrict__ sqy,
    float* __restrict__ acc) {
  __shared__ __align__(16) unsigned short Xi2[2][TILE][LDSW];
  __shared__ __align__(16) unsigned short Xn2[2][TILE][LDSW];
  __shared__ __align__(16) unsigned short Yi2[2][TILE][LDSW];
  __shared__ __align__(16) unsigned short Yn2[2][TILE][LDSW];
  __shared__ float sxi[TILE], sxn[TILE], syi[TILE], syn[TILE];

  // decode upper-triangle tile (iTile <= nTile)
  int iTile = 0, nTile = 0, rem = blockIdx.x;
  for (int it = 0; it < NTILES; ++it) {
    int len = NTILES - it;
    if (rem < len) { iTile = it; nTile = it + rem; break; }
    rem -= len;
  }
  const int iBase = iTile * TILE, nBase = nTile * TILE;
  const bool offDiag = (iTile != nTile);

  const int tid  = threadIdx.x;
  const int lane = tid & 31;
  const int w    = tid >> 5;      // 8 waves
  const int wm   = w & 3;         // 4 M-groups of 32 rows
  const int wn   = w >> 2;        // 2 N-groups of 64 cols
  const int h    = lane >> 4;
  const int lc   = lane & 15;

  if (tid < TILE) {
    sxi[tid] = sqx[iBase + tid]; sxn[tid] = sqx[nBase + tid];
    syi[tid] = sqy[iBase + tid]; syn[tid] = sqy[nBase + tid];
  }

  v8f gx[2][4], gy[2][4];
#pragma unroll
  for (int m = 0; m < 2; ++m)
#pragma unroll
    for (int n = 0; n < 4; ++n) { gx[m][n] = (v8f){}; gy[m][n] = (v8f){}; }

  const unsigned short* xi_g = xb + (size_t)iBase * D_DIM;
  const unsigned short* xn_g = xb + (size_t)nBase * D_DIM;
  const unsigned short* yi_g = yb + (size_t)iBase * D_DIM;
  const unsigned short* yn_g = yb + (size_t)nBase * D_DIM;

  // preload K-chunk 0 into buffer 0 (TDM, wave 0 only; EXEC is ignored by TDM)
  if (w == 0) {
    tdm_load_tile(xi_g, lds_off_of(&Xi2[0][0][0]));
    tdm_load_tile(xn_g, lds_off_of(&Xn2[0][0][0]));
    tdm_load_tile(yi_g, lds_off_of(&Yi2[0][0][0]));
    tdm_load_tile(yn_g, lds_off_of(&Yn2[0][0][0]));
    __builtin_amdgcn_s_wait_tensorcnt(0);
  }
  __syncthreads();

  int p = 0;
  for (int kb = 0; kb < D_DIM; kb += TK) {
    // prefetch next K-chunk into the other buffer while we compute on this one
    if (w == 0 && (kb + TK) < D_DIM) {
      const int q = p ^ 1, ko = kb + TK;
      tdm_load_tile(xi_g + ko, lds_off_of(&Xi2[q][0][0]));
      tdm_load_tile(xn_g + ko, lds_off_of(&Xn2[q][0][0]));
      tdm_load_tile(yi_g + ko, lds_off_of(&Yi2[q][0][0]));
      tdm_load_tile(yn_g + ko, lds_off_of(&Yn2[q][0][0]));
    }

    v16bf ax[2], ay[2];
#pragma unroll
    for (int m = 0; m < 2; ++m) {
      const int r = wm * 32 + m * 16 + lc;
      ax[m] = lds_a_frag(Xi2[p], r, h);
      ay[m] = lds_a_frag(Yi2[p], r, h);
    }
#pragma unroll
    for (int n = 0; n < 4; ++n) {
      const int cb = wn * 64 + n * 16 + lc;
      v16bf bx = lds_b_frag(Xn2[p], cb, h);
      v16bf by = lds_b_frag(Yn2[p], cb, h);
#pragma unroll
      for (int m = 0; m < 2; ++m) {
        gx[m][n] = __builtin_amdgcn_wmma_f32_16x16x32_bf16(
            false, ax[m], false, bx, (short)0, gx[m][n], false, false);
        gy[m][n] = __builtin_amdgcn_wmma_f32_16x16x32_bf16(
            false, ay[m], false, by, (short)0, gy[m][n], false, false);
      }
    }

    // wave 0: DMA for next buffer must be done before anyone reads it
    if (w == 0) __builtin_amdgcn_s_wait_tensorcnt(0);
    __syncthreads();           // also orders this buffer's frag reads vs overwrite
    p ^= 1;
  }

  // ---- epilogue: t = (sxi+sxn-2Gx)*(syi+syn-2Gy); reduce rows (+cols if offDiag)
  float rowsum[2][8];
  float colsum[4] = {0.f, 0.f, 0.f, 0.f};
#pragma unroll
  for (int m = 0; m < 2; ++m)
#pragma unroll
    for (int v = 0; v < 8; ++v) rowsum[m][v] = 0.f;

#pragma unroll
  for (int m = 0; m < 2; ++m) {
#pragma unroll
    for (int n = 0; n < 4; ++n) {
      const int nl = wn * 64 + n * 16 + lc;
      const float sxnv = sxn[nl], synv = syn[nl];
#pragma unroll
      for (int v = 0; v < 8; ++v) {
        const int il = wm * 32 + m * 16 + v + 8 * h;   // C layout: M = v + 8*(lane/16)
        const float tx = sxi[il] + sxnv - 2.0f * gx[m][n][v];
        const float ty = syi[il] + synv - 2.0f * gy[m][n][v];
        const float tt = tx * ty;
        rowsum[m][v] += tt;
        colsum[n]    += tt;
      }
    }
  }

#pragma unroll
  for (int m = 0; m < 2; ++m)
#pragma unroll
    for (int v = 0; v < 8; ++v) {
      float rs = rowsum[m][v];
      rs += __shfl_xor(rs, 8, 16);
      rs += __shfl_xor(rs, 4, 16);
      rs += __shfl_xor(rs, 2, 16);
      rs += __shfl_xor(rs, 1, 16);
      if (lc == 0)
        atomicAdd(&acc[iBase + wm * 32 + m * 16 + v + 8 * h], rs);
    }

  if (offDiag) {
#pragma unroll
    for (int n = 0; n < 4; ++n) {
      float cs = colsum[n];
      cs += __shfl_xor(cs, 16, 32);   // fold the two 8-row halves
      if (h == 0)
        atomicAdd(&acc[nBase + wn * 64 + n * 16 + lc], cs);
    }
  }
}

// --------------------------------- finalize ---------------------------------
__global__ __launch_bounds__(256) void finalize_kernel(const float* __restrict__ acc,
                                                       float* __restrict__ out) {
  const int i = blockIdx.x * 256 + threadIdx.x;
  if (i < B_DIM) out[i] = -acc[i] * (1.0f / ((float)D_DIM * (float)D_DIM));
}

// ---------------------------------------------------------------------------
extern "C" void kernel_launch(void* const* d_in, const int* in_sizes, int n_in,
                              void* d_out, int out_size, void* d_ws, size_t ws_size,
                              hipStream_t stream) {
  const float* x = (const float*)d_in[0];
  const float* y = (const float*)d_in[1];
  float* out = (float*)d_out;

  char* ws = (char*)d_ws;
  const size_t bfBytes = (size_t)B_DIM * D_DIM * 2;   // 32 MB each
  unsigned short* xb = (unsigned short*)ws;
  unsigned short* yb = (unsigned short*)(ws + bfBytes);
  float* sqx = (float*)(ws + 2 * bfBytes);
  float* sqy = sqx + B_DIM;
  float* acc = sqy + B_DIM;

  prep_kernel<<<B_DIM, 256, 0, stream>>>(x, y, xb, yb, sqx, sqy, acc);
  gram_kernel<<<NTRI, 256, 0, stream>>>(xb, yb, sqx, sqy, acc);
  finalize_kernel<<<(B_DIM + 255) / 256, 256, 0, stream>>>(acc, out);
}
